// RPN_79130477462044
// MI455X (gfx1250) — matrix operand order repaired and checked by reading, not compiled, not run on hardware
//
#include <hip/hip_runtime.h>
#include <hip/hip_bf16.h>

typedef __attribute__((ext_vector_type(16))) __bf16 v16bf;
typedef __attribute__((ext_vector_type(8)))  __bf16 v8bf;
typedef __attribute__((ext_vector_type(8)))  float  v8f;

#define IC 256

// ---------------------------------------------------------------------------
// Zero page init (border taps read from here -> no exec-masked loads)
// ---------------------------------------------------------------------------
__global__ void rpn_zero_page(__bf16* zp) {
    zp[threadIdx.x] = (__bf16)0.f;
}

// ---------------------------------------------------------------------------
// Weight prep: OIHW f32 -> [tap][OCp][256] bf16 (pad oc rows >= OCsrc with 0)
// ---------------------------------------------------------------------------
__global__ void rpn_prep_w(const float* __restrict__ src, __bf16* __restrict__ dst,
                           int OCp, int OCsrc, int ICt) {
    int idx = blockIdx.x * blockDim.x + threadIdx.x;
    int total = 9 * OCp * IC;
    if (idx >= total) return;
    int ic  = idx % IC;
    int t   = idx / IC;
    int oc  = t % OCp;
    int tap = t / OCp;
    float v = 0.f;
    if (oc < OCsrc) {
        int ky = tap / 3, kx = tap % 3;
        v = src[(((size_t)oc * ICt + ic) * 3 + ky) * 3 + kx];
    }
    dst[idx] = (__bf16)v;
}

// ---------------------------------------------------------------------------
// NCHW f32 -> [b][p][c] bf16 (LDS-tiled transpose). C == 256 exactly.
// ---------------------------------------------------------------------------
__global__ void rpn_nchw_to_pc(const float* __restrict__ x, __bf16* __restrict__ out, int P) {
    __shared__ float tile[32][33];
    int b = blockIdx.z;
    int pBase = blockIdx.x * 32;
    int cBase = blockIdx.y * 32;
    int tx = threadIdx.x, ty = threadIdx.y;     // 32 x 8
    const float* xb = x + (size_t)b * IC * P;
#pragma unroll
    for (int j = 0; j < 4; ++j) {
        int c = cBase + ty + j * 8;
        int p = pBase + tx;
        tile[ty + j * 8][tx] = (p < P) ? xb[(size_t)c * P + p] : 0.f;
    }
    __syncthreads();
    __bf16* ob = out + (size_t)b * P * IC;
#pragma unroll
    for (int j = 0; j < 4; ++j) {
        int p = pBase + ty + j * 8;
        int c = cBase + tx;
        if (p < P) ob[(size_t)p * IC + c] = (__bf16)tile[tx][ty + j * 8];
    }
}

// ---------------------------------------------------------------------------
// Implicit-GEMM conv3x3 (pad=1) via WMMA bf16 -> f32, 2x2 register blocking.
// One wave computes a 32(oc) x 32(pos) tile (HAS_OC1=true) or a 16x32 tile
// (HAS_OC1=false, used for the 16-row-padded qy conv). Each fragment load
// feeds two WMMAs. Border taps read from the zero page via address select
// (v_cndmask) -- no exec-masked loads, single basic block per tap body.
//   xT : [b][p][256] bf16   Wb : [tap][OCp][256] bf16   out: [b][p][OCs] bf16
// A frag (16x32): lane m=lane%16, kb=(lane<16?0:16), halfwords K=[kb,kb+16)
// B frag (32x16): lane n=lane%16, kbase=(lane<16?0:16), halfwords K=[kbase,kbase+16)
// D (16x16 f32): lane n=lane%16, vgpr r -> m = r + (lane<16?0:8)
// ---------------------------------------------------------------------------
template <bool HAS_OC1>
__global__ __launch_bounds__(32)
void rpn_conv3x3_wmma(const __bf16* __restrict__ xT, const __bf16* __restrict__ Wb,
                      const float* __restrict__ bias, int biasN,
                      const __bf16* __restrict__ zpage,
                      __bf16* __restrict__ out, int OCp, int OCs,
                      int H, int W, int P, int relu) {
    const int lane = threadIdx.x & 31;
    const int b    = blockIdx.z;
    const int n    = lane & 15;          // position within tile / oc row within tile
    const int m    = n;
    const int kbB  = (lane < 16) ? 0 : 16;
    const int kbA  = (lane < 16) ? 0 : 16;

    const int oc0 = blockIdx.y * 32;
    const int oc1 = oc0 + 16;

    const int pA = blockIdx.x * 32 + n;
    const int pB = pA + 16;
    const int hA = pA / W, wA = pA % W;
    const int hB = pB / W, wB = pB % W;
    const bool pAv = (pA < P), pBv = (pB < P);

    v8f acc00 = {}, acc01 = {}, acc10 = {}, acc11 = {};
    const __bf16* xb = xT + (size_t)b * P * IC;
    const __bf16* zp = zpage + kbB;

    for (int tap = 0; tap < 9; ++tap) {
        const int dy = tap / 3 - 1, dx = tap % 3 - 1;
        const int hsA = hA + dy, wsA = wA + dx;
        const int hsB = hB + dy, wsB = wB + dx;
        const bool vA = pAv && hsA >= 0 && hsA < H && wsA >= 0 && wsA < W;
        const bool vB = pBv && hsB >= 0 && hsB < H && wsB >= 0 && wsB < W;
        const __bf16* srcA = vA ? (xb + (size_t)(hsA * W + wsA) * IC + kbB) : zp;
        const __bf16* srcB = vB ? (xb + (size_t)(hsB * W + wsB) * IC + kbB) : zp;
        const __bf16* w0 = Wb + ((size_t)tap * OCp + oc0 + m) * IC + kbA;
        const __bf16* w1 = w0 + (size_t)16 * IC;

#pragma unroll
        for (int kc = 0; kc < 8; ++kc) {
            v8bf a0lo = *(const v8bf*)(w0 + kc * 32);
            v8bf a0hi = *(const v8bf*)(w0 + kc * 32 + 8);
            v8bf bAlo = *(const v8bf*)(srcA + kc * 32);
            v8bf bAhi = *(const v8bf*)(srcA + kc * 32 + 8);
            v8bf bBlo = *(const v8bf*)(srcB + kc * 32);
            v8bf bBhi = *(const v8bf*)(srcB + kc * 32 + 8);
            v16bf a0f, bAf, bBf;
#pragma unroll
            for (int i = 0; i < 8; ++i) {
                a0f[i] = a0lo[i]; a0f[8 + i] = a0hi[i];
                bAf[i] = bAlo[i]; bAf[8 + i] = bAhi[i];
                bBf[i] = bBlo[i]; bBf[8 + i] = bBhi[i];
            }
            acc00 = __builtin_amdgcn_wmma_f32_16x16x32_bf16(
                false, a0f, false, bAf, (short)0, acc00, false, false);
            acc01 = __builtin_amdgcn_wmma_f32_16x16x32_bf16(
                false, a0f, false, bBf, (short)0, acc01, false, false);
            if constexpr (HAS_OC1) {
                v8bf a1lo = *(const v8bf*)(w1 + kc * 32);
                v8bf a1hi = *(const v8bf*)(w1 + kc * 32 + 8);
                v16bf a1f;
#pragma unroll
                for (int i = 0; i < 8; ++i) { a1f[i] = a1lo[i]; a1f[8 + i] = a1hi[i]; }
                acc10 = __builtin_amdgcn_wmma_f32_16x16x32_bf16(
                    false, a1f, false, bAf, (short)0, acc10, false, false);
                acc11 = __builtin_amdgcn_wmma_f32_16x16x32_bf16(
                    false, a1f, false, bBf, (short)0, acc11, false, false);
            }
        }
    }

    // epilogue: bias (+relu) and store; D layout -> lane holds 8 consecutive oc
    const int half = (lane < 16) ? 0 : 8;
    const int nq = HAS_OC1 ? 4 : 2;
#pragma unroll
    for (int q = 0; q < nq; ++q) {
        const bool isOc1 = (q & 2);
        const bool isPB = (q & 1);
        const int p = isPB ? pB : pA;
        if (p >= P) continue;
        const v8f* accp = (q == 0) ? &acc00 : (q == 1) ? &acc01 : (q == 2) ? &acc10 : &acc11;
        const int ocbase = (isOc1 ? oc1 : oc0) + half;
        v8bf st;
#pragma unroll
        for (int r = 0; r < 8; ++r) {
            int oc = ocbase + r;
            float f = (*accp)[r] + ((oc < biasN) ? bias[oc] : 0.f);
            if (relu) f = fmaxf(f, 0.f);
            st[r] = (__bf16)f;
        }
        *(v8bf*)(out + ((size_t)b * P + p) * OCs + ocbase) = st;
    }
}

// ---------------------------------------------------------------------------
// Epilogue: per position -> softmax(qy), 4 GMM components via one-hot-constant
// trick, 1x1 cls/mean heads, anchor decode, clip, write [B, totalP, 6].
// One wave per position; lanes parallel over 256 channels.
// ---------------------------------------------------------------------------
__global__ __launch_bounds__(128)
void rpn_epilogue(const __bf16* __restrict__ u, const __bf16* __restrict__ qy,
                  const float* __restrict__ w_rpn, const float* __restrict__ w_cls,
                  const float* __restrict__ b_cls, const float* __restrict__ w_mean,
                  const float* __restrict__ b_mean, const float* __restrict__ im_info,
                  float* __restrict__ out, int H, int W, int P, float stride,
                  int levelOff, int totalP) {
    const int lane = threadIdx.x & 31;
    const int pos  = blockIdx.x * 4 + (threadIdx.x >> 5);
    const int b    = blockIdx.z;
    if (pos >= P) return;
    const int hh = pos / W, ww = pos % W;

    bool tv[9];
#pragma unroll
    for (int t = 0; t < 9; ++t) {
        int hs = hh + t / 3 - 1, ws2 = ww + t % 3 - 1;
        tv[t] = (hs >= 0) && (hs < H) && (ws2 >= 0) && (ws2 < W);
    }

    const __bf16* qp = qy + ((size_t)b * P + pos) * 16;
    float q0 = (float)qp[0], q1 = (float)qp[1], q2 = (float)qp[2], q3 = (float)qp[3];
    float qm = fmaxf(fmaxf(q0, q1), fmaxf(q2, q3));
    float e0 = __expf(q0 - qm), e1 = __expf(q1 - qm), e2 = __expf(q2 - qm), e3 = __expf(q3 - qm);
    float inv = 1.f / (e0 + e1 + e2 + e3);
    float mix[4] = {e0 * inv, e1 * inv, e2 * inv, e3 * inv};

    const __bf16* up = u + ((size_t)b * P + pos) * IC;
    float bb0 = 0.f, bb1 = 0.f, bb2 = 0.f, bb3 = 0.f, s0 = 0.f, s1 = 0.f;

    for (int nn = 0; nn < 4; ++nn) {
        float d0 = 0.f, d1 = 0.f, d2 = 0.f, d3 = 0.f, c0 = 0.f, c1 = 0.f;
        for (int c = lane; c < IC; c += 32) {
            float uc = (float)up[c];
            // one-hot channel (256+nn) 3x3 tap weights for output channel c
            const float* wr = w_rpn + (size_t)c * (260 * 9) + (size_t)(256 + nn) * 9;
            float oh = 0.f;
#pragma unroll
            for (int t = 0; t < 9; ++t) if (tv[t]) oh += wr[t];
            float tval = fmaxf(uc + oh, 0.f);
            d0 += w_mean[c]       * tval;
            d1 += w_mean[256 + c] * tval;
            d2 += w_mean[512 + c] * tval;
            d3 += w_mean[768 + c] * tval;
            c0 += w_cls[c]        * tval;
            c1 += w_cls[256 + c]  * tval;
        }
#pragma unroll
        for (int o = 16; o >= 1; o >>= 1) {
            d0 += __shfl_xor(d0, o, 32);
            d1 += __shfl_xor(d1, o, 32);
            d2 += __shfl_xor(d2, o, 32);
            d3 += __shfl_xor(d3, o, 32);
            c0 += __shfl_xor(c0, o, 32);
            c1 += __shfl_xor(c1, o, 32);
        }
        float mw = mix[nn];
        bb0 += mw * (d0 + b_mean[0]);
        bb1 += mw * (d1 + b_mean[1]);
        bb2 += mw * (d2 + b_mean[2]);
        bb3 += mw * (d3 + b_mean[3]);
        s0  += mw * (c0 + b_cls[0]);
        s1  += mw * (c1 + b_cls[1]);
    }

    if (lane == 0) {
        float aw  = stride * 8.0f;
        float acx = (ww + 0.5f) * stride;
        float acy = (hh + 0.5f) * stride;
        float cx = acx + bb0 * aw;
        float cy = acy + bb1 * aw;
        float bw = aw * __expf(fminf(fmaxf(bb2, -10.f), 10.f));
        float bh = aw * __expf(fminf(fmaxf(bb3, -10.f), 10.f));
        float x1 = cx - bw * 0.5f, y1 = cy - bh * 0.5f;
        float x2 = cx + bw * 0.5f, y2 = cy + bh * 0.5f;
        float limx = im_info[b * 6 + 1], limy = im_info[b * 6 + 0];
        x1 = fminf(fmaxf(x1, 0.f), limx);
        y1 = fminf(fmaxf(y1, 0.f), limy);
        x2 = fminf(fmaxf(x2, 0.f), limx);
        y2 = fminf(fmaxf(y2, 0.f), limy);
        float* op = out + ((size_t)b * totalP + levelOff + pos) * 6;
        op[0] = x1; op[1] = y1; op[2] = x2; op[3] = y2; op[4] = s0; op[5] = s1;
    }
}

// ---------------------------------------------------------------------------
// Host launcher
// ---------------------------------------------------------------------------
extern "C" void kernel_launch(void* const* d_in, const int* in_sizes, int n_in,
                              void* d_out, int out_size, void* d_ws, size_t ws_size,
                              hipStream_t stream) {
    const float* feats[5] = {(const float*)d_in[0], (const float*)d_in[1],
                             (const float*)d_in[2], (const float*)d_in[3],
                             (const float*)d_in[4]};
    const float* im_info = (const float*)d_in[5];
    const float* w_qy1 = (const float*)d_in[6];
    const float* b_qy1 = (const float*)d_in[7];
    const float* w_qy2 = (const float*)d_in[8];
    const float* b_qy2 = (const float*)d_in[9];
    const float* w_qy  = (const float*)d_in[10];
    const float* b_qy  = (const float*)d_in[11];
    const float* w_rpn = (const float*)d_in[12];
    const float* b_rpn = (const float*)d_in[13];
    const float* w_cls = (const float*)d_in[14];
    const float* b_cls = (const float*)d_in[15];
    const float* w_mean = (const float*)d_in[16];
    const float* b_mean = (const float*)d_in[17];

    const int Hl[5] = {13, 25, 50, 100, 200};
    const int Wl[5] = {19, 38, 76, 152, 304};
    const float strides[5] = {64.f, 32.f, 16.f, 8.f, 4.f};
    int totalP = 0;
    for (int l = 0; l < 5; ++l) totalP += Hl[l] * Wl[l];
    const int Pmax = Hl[4] * Wl[4];   // 60800

    // workspace carve-out
    char* ws = (char*)d_ws;
    size_t off = 0;
    auto carve = [&](size_t bytes) -> void* {
        bytes = (bytes + 255) & ~(size_t)255;
        void* p = ws + off;
        off += bytes;
        return p;
    };
    __bf16* zpage = (__bf16*)carve(256 * 2);
    __bf16* Wq1  = (__bf16*)carve((size_t)9 * 256 * IC * 2);
    __bf16* Wq2  = (__bf16*)carve((size_t)9 * 256 * IC * 2);
    __bf16* Wrpn = (__bf16*)carve((size_t)9 * 256 * IC * 2);
    __bf16* Wqy  = (__bf16*)carve((size_t)9 * 16  * IC * 2);
    size_t actBytes = (size_t)2 * Pmax * IC * 2;   // [B][P][256] bf16
    __bf16* bufA = (__bf16*)carve(actBytes);
    __bf16* bufB = (__bf16*)carve(actBytes);
    __bf16* bufC = (__bf16*)carve(actBytes);
    __bf16* qybuf = (__bf16*)carve((size_t)2 * Pmax * 16 * 2);
    (void)ws_size;

    // --- one-time per launch: zero page + weight prep ---
    rpn_zero_page<<<1, 256, 0, stream>>>(zpage);
    {
        int tot = 9 * 256 * IC;
        rpn_prep_w<<<(tot + 255) / 256, 256, 0, stream>>>(w_qy1, Wq1, 256, 256, 256);
        rpn_prep_w<<<(tot + 255) / 256, 256, 0, stream>>>(w_qy2, Wq2, 256, 256, 256);
        rpn_prep_w<<<(tot + 255) / 256, 256, 0, stream>>>(w_rpn, Wrpn, 256, 256, 260);
        int tot2 = 9 * 16 * IC;
        rpn_prep_w<<<(tot2 + 255) / 256, 256, 0, stream>>>(w_qy, Wqy, 16, 4, 256);
    }

    int levelOff = 0;
    for (int l = 0; l < 5; ++l) {
        const int H = Hl[l], W = Wl[l], P = H * W;
        const int posTiles = (P + 31) / 32;

        // x (NCHW f32) -> bufA ([b][p][c] bf16)
        rpn_nchw_to_pc<<<dim3((P + 31) / 32, IC / 32, 2), dim3(32, 8), 0, stream>>>(
            feats[l], bufA, P);

        // h1 = relu(conv(x, w_qy1) + b_qy1)
        rpn_conv3x3_wmma<true><<<dim3(posTiles, 8, 2), 32, 0, stream>>>(
            bufA, Wq1, b_qy1, 256, zpage, bufB, 256, 256, H, W, P, 1);
        // h2 = relu(conv(h1, w_qy2) + b_qy2)
        rpn_conv3x3_wmma<true><<<dim3(posTiles, 8, 2), 32, 0, stream>>>(
            bufB, Wq2, b_qy2, 256, zpage, bufC, 256, 256, H, W, P, 1);
        // qy = conv(h2, w_qy) + b_qy   (16-row padded tile, rows 4..15 are zero)
        rpn_conv3x3_wmma<false><<<dim3(posTiles, 1, 2), 32, 0, stream>>>(
            bufC, Wqy, b_qy, 4, zpage, qybuf, 16, 16, H, W, P, 0);
        // u = conv(x, w_rpn[:, :256]) + b_rpn   (pre-relu, shared across components)
        rpn_conv3x3_wmma<true><<<dim3(posTiles, 8, 2), 32, 0, stream>>>(
            bufA, Wrpn, b_rpn, 256, zpage, bufB, 256, 256, H, W, P, 0);

        // GMM combine + decode + clip -> d_out
        rpn_epilogue<<<dim3((P + 3) / 4, 1, 2), 128, 0, stream>>>(
            bufB, qybuf, w_rpn, w_cls, b_cls, w_mean, b_mean, im_info,
            (float*)d_out, H, W, P, strides[l], levelOff, totalP);

        levelOff += P;
    }
}
